// GINE_86182813761570
// MI455X (gfx1250) — matrix operand order repaired and checked by reading, not compiled, not run on hardware
//
#include <hip/hip_runtime.h>
#include <hip/hip_bf16.h>

typedef __attribute__((ext_vector_type(16))) _Float16 v16h;
typedef __attribute__((ext_vector_type(8)))  _Float16 v8h;
typedef __attribute__((ext_vector_type(8)))  float    v8f;
typedef __attribute__((ext_vector_type(4)))  unsigned int v4u;
typedef __attribute__((ext_vector_type(8)))  int      v8i;
typedef __attribute__((ext_vector_type(4)))  int      v4i;

#define DIM       256
#define N_NODES   10000
#define N_EDGES   320000
#define EDGE_TILES (N_EDGES / 16)   // 20000
#define NODE_TILES (N_NODES / 16)   // 625
#define FRAG_ELEMS (256 * 256)      // one 256x256 matrix as f16 fragments
#define FRAG_BYTES (FRAG_ELEMS * 2) // 131072
#define EDGE_BLOCKS 500             // 8 waves/block * 5 tiles/wave = 40 tiles/block

// ---------------------------------------------------------------------------
// Pack a row-major f32 256x256 weight W[k][n] into WMMA B-fragment order, f16.
// Fragment index: ((ks*16 + nt)*32 + lane)*16 + i
//   column n = nt*16 + (lane&15), k = ks*32 + (lane>>4)*8 + (i<8 ? i : i+8)
// ---------------------------------------------------------------------------
__global__ void pack_bfrag_kernel(const float* __restrict__ W,
                                  _Float16* __restrict__ frag) {
    int idx  = blockIdx.x * blockDim.x + threadIdx.x;   // 0..65535
    int i    = idx & 15;
    int lane = (idx >> 4) & 31;
    int t    = idx >> 9;
    int nt   = t & 15;
    int ks   = t >> 4;
    int k = ks * 32 + (lane >> 4) * 8 + (i < 8 ? i : i + 8);
    int n = nt * 16 + (lane & 15);
    frag[idx] = (_Float16)W[k * DIM + n];
}

__global__ void zero_kernel(float* __restrict__ p, int n) {
    int idx = blockIdx.x * blockDim.x + threadIdx.x;
    if (idx < n) p[idx] = 0.0f;
}

__device__ __forceinline__ v8f wmma_f16(v16h a, v16h b, v8f c) {
    return __builtin_amdgcn_wmma_f32_16x16x32_f16(
        /*neg_a=*/false, a, /*neg_b=*/false, b,
        /*c_mod=*/(short)0, c, /*reuse_a=*/false, /*reuse_b=*/false);
}

// ---------------------------------------------------------------------------
// Stage one fragment-ordered 256x256 f16 matrix (128 KB) into LDS.
// Preferred path: Tensor Data Mover (TENSOR_LOAD_TO_LDS), issued by wave 0
// (TDM ignores EXEC; one issue per wave). Fallback: cooperative vector copy.
// D# describes a 1-D tile of 16384 x 8-byte elements.
// ---------------------------------------------------------------------------
__device__ __forceinline__ void stage_frag_to_lds(const _Float16* __restrict__ g,
                                                  _Float16* lds) {
#if __has_builtin(__builtin_amdgcn_tensor_load_to_lds)
    if ((threadIdx.x >> 5) == 0) {
        unsigned long long ga = (unsigned long long)g;
        unsigned int       la = (unsigned int)(unsigned long long)lds;
        v4u g0;
        g0[0] = 1u;                                        // count=1, no gather
        g0[1] = la;                                        // lds_addr
        g0[2] = (unsigned int)ga;                          // global_addr[31:0]
        g0[3] = ((unsigned int)(ga >> 32) & 0x01FFFFFFu)   // global_addr[56:32]
                | (2u << 30);                              // type=2 (image)
        v8i g1;
        g1[0] = (int)(3u << 16);       // workgroup_mask=0, data_size=3 (8B)
        g1[1] = (int)(0x4000u << 16);  // tensor_dim0[15:0]=16384 in bits[63:48]
        g1[2] = (int)(1u << 16);       // tensor_dim0[31:16]=0, tensor_dim1=1
        g1[3] = (int)(0x4000u << 16);  // tile_dim0=16384 in bits[127:112]
        g1[4] = 0;                     // tile_dim1=0 (unused), tile_dim2=0
        g1[5] = 16384;                 // tensor_dim0_stride lo
        g1[6] = (int)(0x4000u << 16);  // stride0 hi=0, tensor_dim1_stride lo16
        g1[7] = 0;
        v4i gz = {0, 0, 0, 0};
#if __clang_major__ >= 23
        v8i z8 = {0, 0, 0, 0, 0, 0, 0, 0};
        __builtin_amdgcn_tensor_load_to_lds(g0, g1, gz, gz, z8, 0);
#else
        __builtin_amdgcn_tensor_load_to_lds(g0, g1, gz, gz, 0);
#endif
#if __has_builtin(__builtin_amdgcn_s_wait_tensorcnt)
        __builtin_amdgcn_s_wait_tensorcnt(0);
#else
        asm volatile("s_wait_tensorcnt 0x0" ::: "memory");
#endif
    }
    __syncthreads();
#else
    const float4* gs = (const float4*)g;
    float4*       ld = (float4*)lds;
    for (int i = threadIdx.x; i < FRAG_BYTES / 16; i += blockDim.x)
        ld[i] = gs[i];
    __syncthreads();
#endif
}

// ---------------------------------------------------------------------------
// Edge kernel: 8 waves/block, 5 tiles of 16 edges per wave.
// W2 fragments staged once per block into LDS (TDM), B operands via ds_load.
//   H[e][k]   = relu(PE[e]*W1[k] + b1[k])           (A fragments, in regs)
//   P         = H @ W2 + b2                          (WMMA)
//   msg[e][n] = relu(X[src][n] + emb[attr][n]*P[e][n])
//   S[dst]   += msg                                  (global f32 atomics)
// ---------------------------------------------------------------------------
__global__ void __launch_bounds__(256)
edge_kernel(const float* __restrict__ X,
            const int* __restrict__ src,
            const int* __restrict__ dst,
            const int* __restrict__ attr,
            const float* __restrict__ PE,
            const float* __restrict__ emb,
            const float* __restrict__ peW1,
            const float* __restrict__ peb1,
            const _Float16* __restrict__ fragW2,
            const float* __restrict__ peb2,
            float* __restrict__ S) {
    extern __shared__ _Float16 bsm[];   // 65536 f16 = 128 KB fragment cache

    stage_frag_to_lds(fragW2, bsm);

    int wave = threadIdx.x >> 5;
    int lane = threadIdx.x & 31;
    int m    = lane & 15;
    int hi   = lane >> 4;

    for (int t = 0; t < 5; ++t) {
        int tile = blockIdx.x * 40 + wave * 5 + t;

        // ---- A fragments: lane owns edge row m, specific k-pattern ---------
        float pe = PE[tile * 16 + m];
        v16h a[8];
#pragma unroll
        for (int ks = 0; ks < 8; ++ks) {
#pragma unroll
            for (int i = 0; i < 16; ++i) {
                int k = ks * 32 + hi * 8 + (i < 8 ? i : i + 8);
                float h = pe * peW1[k] + peb1[k];
                a[ks][i] = (_Float16)(h > 0.0f ? h : 0.0f);
            }
        }

        // ---- per-lane metadata for its 8 C-rows (M = v + hi*8) -------------
        int s8[8], d8[8], t8[8];
#pragma unroll
        for (int v = 0; v < 8; ++v) {
            int e = tile * 16 + v + hi * 8;
            s8[v] = src[e];
            d8[v] = dst[e];
            t8[v] = attr[e] - 1;
            __builtin_prefetch(&X[(long)s8[v] * DIM], 0, 1);  // global_prefetch
        }

        // ---- 16 N-tiles of the output row-block ----------------------------
        for (int nt = 0; nt < 16; ++nt) {
            int n = nt * 16 + m;                  // C layout: lane column
            float bv = peb2[n];
            v8f c = {bv, bv, bv, bv, bv, bv, bv, bv};
#pragma unroll
            for (int ks = 0; ks < 8; ++ks) {
                const _Float16* bp = bsm + (((ks * 16 + nt) * 32 + lane) << 4);
                v16h b = *(const v16h*)bp;        // 2x ds_load_b128
                c = wmma_f16(a[ks], b, c);
            }
#pragma unroll
            for (int v = 0; v < 8; ++v) {
                float xe  = emb[t8[v] * DIM + n] * c[v];
                float msg = X[(long)s8[v] * DIM + n] + xe;
                msg = msg > 0.0f ? msg : 0.0f;
                unsafeAtomicAdd(&S[(long)d8[v] * DIM + n], msg);  // global_atomic_add_f32
            }
        }
    }
}

// ---------------------------------------------------------------------------
// Node kernel: one wave per tile of 16 nodes.
//   Z = (1+eps)*X + S
//   H = relu(Z @ W1 + b1)   -> staged f16 in LDS
//   O = H @ W2 + b2
// ---------------------------------------------------------------------------
__global__ void __launch_bounds__(32)
node_kernel(const float* __restrict__ X,
            const float* __restrict__ S,
            const float* __restrict__ eps,
            const _Float16* __restrict__ fragW1,
            const float* __restrict__ b1,
            const _Float16* __restrict__ fragW2,
            const float* __restrict__ b2,
            float* __restrict__ Xout) {
    __shared__ _Float16 Hl[16 * DIM];           // 8 KB per (single-wave) block
    int tile = blockIdx.x;
    if (tile >= NODE_TILES) return;
    int lane = threadIdx.x & 31;
    int m    = lane & 15;
    int hi   = lane >> 4;
    float ep = 1.0f + eps[0];

    // ---- A fragments of Z --------------------------------------------------
    long row = (long)(tile * 16 + m) * DIM;
    v16h a[8];
#pragma unroll
    for (int ks = 0; ks < 8; ++ks) {
#pragma unroll
        for (int i = 0; i < 16; ++i) {
            int k = ks * 32 + hi * 8 + (i < 8 ? i : i + 8);
            float z = ep * X[row + k] + S[row + k];
            a[ks][i] = (_Float16)z;
        }
    }

    // ---- GEMM1: H = relu(Z @ W1 + b1), stage to LDS ------------------------
    for (int nt = 0; nt < 16; ++nt) {
        int n = nt * 16 + m;
        float bv = b1[n];
        v8f c = {bv, bv, bv, bv, bv, bv, bv, bv};
#pragma unroll
        for (int ks = 0; ks < 8; ++ks) {
            const _Float16* bp = fragW1 + (((ks * 16 + nt) * 32 + lane) << 4);
            v16h b = *(const v16h*)bp;
            c = wmma_f16(a[ks], b, c);
        }
#pragma unroll
        for (int v = 0; v < 8; ++v) {
            float h = c[v];
            Hl[(v + hi * 8) * DIM + n] = (_Float16)(h > 0.0f ? h : 0.0f);
        }
    }
    __syncthreads();  // single-wave block: forces dscnt ordering

    // ---- reload H as A fragments (ds_load_b128 x2 per kstep) ---------------
    v16h a2[8];
#pragma unroll
    for (int ks = 0; ks < 8; ++ks) {
        const _Float16* p0 = &Hl[m * DIM + ks * 32 + hi * 8];
        v8h lo = *(const v8h*)p0;               // 16B LDS load
        v8h hh = *(const v8h*)(p0 + 16);        // 16B LDS load
        a2[ks] = __builtin_shufflevector(lo, hh, 0, 1, 2, 3, 4, 5, 6, 7,
                                                 8, 9, 10, 11, 12, 13, 14, 15);
    }

    // ---- GEMM2: O = H @ W2 + b2 -------------------------------------------
    for (int nt = 0; nt < 16; ++nt) {
        int n = nt * 16 + m;
        float bv = b2[n];
        v8f c = {bv, bv, bv, bv, bv, bv, bv, bv};
#pragma unroll
        for (int ks = 0; ks < 8; ++ks) {
            const _Float16* bp = fragW2 + (((ks * 16 + nt) * 32 + lane) << 4);
            v16h b = *(const v16h*)bp;
            c = wmma_f16(a2[ks], b, c);
        }
#pragma unroll
        for (int v = 0; v < 8; ++v) {
            Xout[(long)(tile * 16 + v + hi * 8) * DIM + n] = c[v];
        }
    }
}

// ---------------------------------------------------------------------------
// Orchestration
// ---------------------------------------------------------------------------
extern "C" void kernel_launch(void* const* d_in, const int* in_sizes, int n_in,
                              void* d_out, int out_size, void* d_ws, size_t ws_size,
                              hipStream_t stream) {
    const float* X_n  = (const float*)d_in[0];
    const int*   ei   = (const int*)d_in[1];      // (2, E): row0=src, row1=dst
    const int*   src  = ei;
    const int*   dst  = ei + N_EDGES;
    const int*   attr = (const int*)d_in[2];
    const float* PE   = (const float*)d_in[3];

    // Workspace layout
    float* buf0 = (float*)d_ws;                         // N*D f32
    float* buf1 = buf0 + (size_t)N_NODES * DIM;         // N*D f32
    float* S    = buf1 + (size_t)N_NODES * DIM;         // N*D f32
    _Float16* fragPeW2 = (_Float16*)(S + (size_t)N_NODES * DIM);
    _Float16* fragW1   = fragPeW2 + FRAG_ELEMS;
    _Float16* fragW2   = fragW1 + FRAG_ELEMS;

    const float* Xin = X_n;
    for (int l = 0; l < 3; ++l) {
        int base = 4 + 10 * l;
        const float* emb   = (const float*)d_in[base + 0];
        const float* eps   = (const float*)d_in[base + 1];
        const float* peW1  = (const float*)d_in[base + 2];
        const float* peb1  = (const float*)d_in[base + 3];
        const float* peW2  = (const float*)d_in[base + 4];
        const float* peb2  = (const float*)d_in[base + 5];
        const float* mW1   = (const float*)d_in[base + 6];
        const float* mb1   = (const float*)d_in[base + 7];
        const float* mW2   = (const float*)d_in[base + 8];
        const float* mb2   = (const float*)d_in[base + 9];

        float* Xout = (l == 2) ? (float*)d_out : ((l == 0) ? buf0 : buf1);

        pack_bfrag_kernel<<<FRAG_ELEMS / 256, 256, 0, stream>>>(peW2, fragPeW2);
        pack_bfrag_kernel<<<FRAG_ELEMS / 256, 256, 0, stream>>>(mW1, fragW1);
        pack_bfrag_kernel<<<FRAG_ELEMS / 256, 256, 0, stream>>>(mW2, fragW2);
        zero_kernel<<<(N_NODES * DIM + 255) / 256, 256, 0, stream>>>(S, N_NODES * DIM);

        edge_kernel<<<EDGE_BLOCKS, 256, FRAG_BYTES, stream>>>(
            Xin, src, dst, attr, PE, emb, peW1, peb1, fragPeW2, peb2, S);

        node_kernel<<<NODE_TILES, 32, 0, stream>>>(
            Xin, S, eps, fragW1, mb1, fragW2, mb2, Xout);

        Xin = Xout;
    }
}
